// SoftContrastiveLoss_36919538876769
// MI455X (gfx1250) — matrix operand order repaired and checked by reading, not compiled, not run on hardware
//
#include <hip/hip_runtime.h>
#include <stdint.h>

typedef __attribute__((ext_vector_type(16))) _Float16 v16h;
typedef __attribute__((ext_vector_type(8)))  _Float16 v8h;
typedef __attribute__((ext_vector_type(4)))  _Float16 v4h;
typedef __attribute__((ext_vector_type(8)))  float    v8f;

#define NROWS 8192
#define DIM   512
#define DIMP  520            // padded LDS row stride for A (halfs)
#define BM    128            // z1 rows per block (8 waves x 16)
#define BLOCK 256
#define NB    4              // 4 column tiles (64 cols) per group
#define GRPCOLS 64
#define CS    4              // column splits across blocks
#define COLS_PER_BLOCK (NROWS / CS)          // 2048
#define NGROUPS (COLS_PER_BLOCK / GRPCOLS)   // 32
#define INV_TAU 10.0f
#define BCHUNKS ((GRPCOLS * DIM / 8) / BLOCK)  // 16 async b128 ops per thread/panel

// ---------------------------------------------------------------------------
// CDNA5 async global->LDS copy (ASYNCcnt path), with synchronous fallback.
// Builtin signature (probe-confirmed): (as1 int4*, as3 int4*, imm off, imm cpol)
// ---------------------------------------------------------------------------
#if defined(__has_builtin)
#if __has_builtin(__builtin_amdgcn_global_load_async_to_lds_b128) && \
    __has_builtin(__builtin_amdgcn_s_wait_asynccnt)
#define HAVE_ASYNC_LDS 1
#endif
#endif
#ifndef HAVE_ASYNC_LDS
#define HAVE_ASYNC_LDS 0
#endif

typedef int v4i_vec __attribute__((vector_size(16)));
typedef __attribute__((address_space(1))) v4i_vec* g_v4i_p;
typedef __attribute__((address_space(3))) v4i_vec* l_v4i_p;

__device__ __forceinline__ void async_copy16(const _Float16* gsrc, _Float16* ldst) {
#if HAVE_ASYNC_LDS
  __builtin_amdgcn_global_load_async_to_lds_b128(
      (g_v4i_p)(gsrc), (l_v4i_p)(ldst), 0, 0);
#else
  *(uint4*)ldst = *(const uint4*)gsrc;
#endif
}

template <int N>
__device__ __forceinline__ void async_wait_le() {
#if HAVE_ASYNC_LDS
  __builtin_amdgcn_s_wait_asynccnt(N);
#endif
}

// ---------------------------------------------------------------------------
// Kernel 1: L2-normalize rows of z (fp32 [8192,512]) -> f16. Wave per row.
// ---------------------------------------------------------------------------
__global__ __launch_bounds__(256) void normalize_f16_kernel(
    const float* __restrict__ z, _Float16* __restrict__ zh) {
  const int wave = threadIdx.x >> 5;
  const int lane = threadIdx.x & 31;
  const int row  = blockIdx.x * 8 + wave;

  const float4* src = (const float4*)(z + (size_t)row * DIM);
  float4 x[4];
  float ss = 0.0f;
#pragma unroll
  for (int j = 0; j < 4; ++j) {
    x[j] = src[lane + j * 32];
    ss += x[j].x * x[j].x + x[j].y * x[j].y + x[j].z * x[j].z + x[j].w * x[j].w;
  }
#pragma unroll
  for (int m = 1; m < 32; m <<= 1) ss += __shfl_xor(ss, m, 32);

  const float scale = 1.0f / fmaxf(sqrtf(ss), 1e-12f);

  v4h* dst = (v4h*)(zh + (size_t)row * DIM);
#pragma unroll
  for (int j = 0; j < 4; ++j) {
    v4h h;
    h.x = (_Float16)(x[j].x * scale);
    h.y = (_Float16)(x[j].y * scale);
    h.z = (_Float16)(x[j].z * scale);
    h.w = (_Float16)(x[j].w * scale);
    dst[lane + j * 32] = h;
  }
}

// ---------------------------------------------------------------------------
// Kernel 2: fused sim/softmax statistics.
// Grid (64 row-blocks, 4 column-splits), 256 threads (8 waves).
// A panel row-major+padded in LDS; B panels double-buffered in LDS in
// *fragment-major* order (scattered during the async copy) so every B read is
// a ds_load_b128 with a pure immediate offset:
//   fragment f=k*NB+t (1 KB): lo half at f*1024 + lane*16, hi at +512.
// ---------------------------------------------------------------------------
__global__ __launch_bounds__(BLOCK) void soft_contrastive_pass_kernel(
    const _Float16* __restrict__ z1h, const _Float16* __restrict__ z2h,
    const int* __restrict__ ids, float* __restrict__ accDw,
    float* __restrict__ accSmw, float* __restrict__ accS2w) {
  __shared__ _Float16 Ash[BM * DIMP];            // ~130 KB
  __shared__ _Float16 Bsh[2][GRPCOLS * DIM];     // 2 x 64 KB, fragment-major

  const int tid  = threadIdx.x;
  const int wave = tid >> 5;
  const int lane = tid & 31;
  const int g    = lane >> 4;        // half-wave group
  const int ln   = lane & 15;
  const int rowBlock = blockIdx.x * BM;
  const int split    = blockIdx.y;
  const int cb0      = split * COLS_PER_BLOCK;

  // ---- stage A panel (async): 128 rows x 512 halfs, padded rows
  {
    const _Float16* a0 = z1h + (size_t)rowBlock * DIM;
#pragma unroll
    for (int j = 0; j < (BM * DIM / 8) / BLOCK; ++j) {   // 32 chunks/thread
      int idx = j * BLOCK + tid;
      int r = idx >> 6, cj = idx & 63;
      async_copy16(a0 + (size_t)r * DIM + cj * 8, Ash + r * DIMP + cj * 8);
    }
  }
  // B staging: scatter 16B chunk (row r, chunk cj) into fragment-major order.
  //   k=cj>>2, g=cj&1, hi=(cj>>1)&1, t=r>>4, ln=r&15
  //   dst halfs = (k*NB+t)*512 + hi*256 + (g*16+ln)*8
  auto stageB = [&](const _Float16* bsrc, _Float16* bdst) {
#pragma unroll
    for (int j = 0; j < BCHUNKS; ++j) {                  // 16 chunks/thread
      int idx = j * BLOCK + tid;
      int r = idx >> 6, cj = idx & 63;
      int f  = (cj >> 2) * NB + (r >> 4);
      int lt = (cj & 1) * 16 + (r & 15);
      int hi = (cj >> 1) & 1;
      async_copy16(bsrc + (size_t)r * DIM + cj * 8,
                   bdst + f * 512 + hi * 256 + lt * 8);
    }
  };
  stageB(z2h + (size_t)cb0 * DIM, Bsh[0]);
#if !HAVE_ASYNC_LDS
  __syncthreads();
#endif

  const int rTile = wave * 16;
  int idrow[8];
#pragma unroll
  for (int v = 0; v < 8; ++v) idrow[v] = ids[rowBlock + rTile + g * 8 + v];

  float accD[8] = {}, accSm[8] = {}, accS2[8] = {};

  // A fragment: lane ln owns row (rTile+ln); K = 32k + 8g + {0..7, 16..23}
  const _Float16* Arow = Ash + (size_t)(rTile + ln) * DIMP + g * 8;

  for (int grp = 0; grp < NGROUPS; ++grp) {
    const int col0 = cb0 + grp * GRPCOLS;

    if (grp + 1 < NGROUPS) {
      stageB(z2h + (size_t)(col0 + GRPCOLS) * DIM, Bsh[(grp + 1) & 1]);
      async_wait_le<BCHUNKS>();     // drain all but the in-flight prefetch
    } else {
      async_wait_le<0>();
    }
    __syncthreads();

    const _Float16* Bfrag = Bsh[grp & 1] + lane * 8;  // lane base (16B units)
    int idc[NB];
#pragma unroll
    for (int t = 0; t < NB; ++t) idc[t] = ids[col0 + t * 16 + ln];

    v8f c[NB];
#pragma unroll
    for (int t = 0; t < NB; ++t) c[t] = (v8f){};

#pragma unroll 4
    for (int k = 0; k < DIM / 32; ++k) {
      v8h alo = *(const v8h*)(Arow + k * 32);          // ds_load_b128 (imm)
      v8h ahi = *(const v8h*)(Arow + k * 32 + 16);     // ds_load_b128 (imm)
      v16h a = __builtin_shufflevector(alo, ahi,
          0, 1, 2, 3, 4, 5, 6, 7, 8, 9, 10, 11, 12, 13, 14, 15);
#pragma unroll
      for (int t = 0; t < NB; ++t) {
        const _Float16* bp = Bfrag + (k * NB + t) * 512;
        v8h blo = *(const v8h*)(bp);                   // ds_load_b128 (imm)
        v8h bhi = *(const v8h*)(bp + 256);             // ds_load_b128 (imm)
        v16h b = __builtin_shufflevector(blo, bhi,
            0, 1, 2, 3, 4, 5, 6, 7, 8, 9, 10, 11, 12, 13, 14, 15);
        c[t] = __builtin_amdgcn_wmma_f32_16x16x32_f16(
            false, a, false, b, (short)0, c[t], false, false);
      }
    }

    // fused epilogue: exp + masked accumulation (sim never touches memory)
#pragma unroll
    for (int t = 0; t < NB; ++t) {
#pragma unroll
      for (int v = 0; v < 8; ++v) {
        float e = __expf(c[t][v] * INV_TAU - 10.0f);   // shift by sim bound
        accD[v] += e;
        if (idrow[v] == idc[t]) accSm[v] += e;
        else                    accS2[v] += e * e;
      }
    }
    __syncthreads();   // all waves done reading this B buffer
  }

  // reduce across the 16 lanes of each half-wave (same rows, different N)
#pragma unroll
  for (int v = 0; v < 8; ++v) {
#pragma unroll
    for (int m = 1; m < 16; m <<= 1) {
      accD[v]  += __shfl_xor(accD[v],  m, 32);
      accSm[v] += __shfl_xor(accSm[v], m, 32);
      accS2[v] += __shfl_xor(accS2[v], m, 32);
    }
  }
  if (ln == 0) {
#pragma unroll
    for (int v = 0; v < 8; ++v) {
      const int row = rowBlock + rTile + g * 8 + v;
      const size_t slot = (size_t)split * NROWS + row;
      accDw[slot]  = accD[v];
      accSmw[slot] = accSm[v];
      accS2w[slot] = accS2[v];
    }
  }
}

// ---------------------------------------------------------------------------
// Kernel 3: deterministic single-block finalize. No atomics anywhere.
// ---------------------------------------------------------------------------
__global__ __launch_bounds__(1024) void finalize_kernel(
    const float* __restrict__ accDw, const float* __restrict__ accSmw,
    const float* __restrict__ accS2w, float* __restrict__ out) {
  __shared__ float red[32];
  float lsum = 0.0f;
  for (int r = threadIdx.x; r < NROWS; r += 1024) {
    float Dv = 0.0f, Sm = 0.0f, S2 = 0.0f;
#pragma unroll
    for (int s = 0; s < CS; ++s) {
      Dv += accDw[(size_t)s * NROWS + r];
      Sm += accSmw[(size_t)s * NROWS + r];
      S2 += accS2w[(size_t)s * NROWS + r];
    }
    const float num = Sm + S2 / Dv;
    lsum += -__logf(num / (Dv + 1e-8f) + 1e-8f);
  }
#pragma unroll
  for (int m = 1; m < 32; m <<= 1) lsum += __shfl_xor(lsum, m, 32);
  if ((threadIdx.x & 31) == 0) red[threadIdx.x >> 5] = lsum;
  __syncthreads();
  if (threadIdx.x < 32) {
    float v = red[threadIdx.x];
#pragma unroll
    for (int m = 1; m < 32; m <<= 1) v += __shfl_xor(v, m, 32);
    if (threadIdx.x == 0) out[0] = v * (1.0f / NROWS);
  }
}

// ---------------------------------------------------------------------------
extern "C" void kernel_launch(void* const* d_in, const int* in_sizes, int n_in,
                              void* d_out, int out_size, void* d_ws, size_t ws_size,
                              hipStream_t stream) {
  (void)in_sizes; (void)n_in; (void)out_size; (void)ws_size;
  const float* z1  = (const float*)d_in[0];
  const float* z2  = (const float*)d_in[1];
  const int*   ids = (const int*)d_in[2];
  float* out = (float*)d_out;

  _Float16* z1h = (_Float16*)d_ws;                         // 8 MB
  _Float16* z2h = z1h + (size_t)NROWS * DIM;               // 8 MB
  float* accDw  = (float*)(z2h + (size_t)NROWS * DIM);     // CS*8192 f32
  float* accSmw = accDw + (size_t)CS * NROWS;
  float* accS2w = accSmw + (size_t)CS * NROWS;

  normalize_f16_kernel<<<NROWS / 8, 256, 0, stream>>>(z1, z1h);
  normalize_f16_kernel<<<NROWS / 8, 256, 0, stream>>>(z2, z2h);
  dim3 grid(NROWS / BM, CS);
  soft_contrastive_pass_kernel<<<grid, BLOCK, 0, stream>>>(
      z1h, z2h, ids, accDw, accSmw, accS2w);
  finalize_kernel<<<1, 1024, 0, stream>>>(accDw, accSmw, accS2w, out);
}
